// GRUDECAYLAYERS_45827301048475
// MI455X (gfx1250) — matrix operand order repaired and checked by reading, not compiled, not run on hardware
//
#include <hip/hip_runtime.h>
#include <hip/hip_bf16.h>

// GRU-decay scan, persistent per-batch-tile kernel using v_wmma_f32_16x16x32_bf16.
// B=256, T=512, F=128, U=512.
// Round 2: software-pipeline the B-fragment streams (prefetch next K-step before WMMA).

typedef __attribute__((ext_vector_type(16))) __bf16 v16bf;
typedef __attribute__((ext_vector_type(8)))  float  v8f;

#define Bsz 256
#define Tsz 512
#define Fsz 128
#define Usz 512

// Workspace layout (bf16 halves), WMMA-B-fragment swizzled:
//  w frags: [ntile=32][kstep=4 ][512 halves]   u frags: [ntile=32][kstep=16][512 halves]
#define WFRAG (32L * 4  * 512)   // 65536 halves per w matrix
#define UFRAG (32L * 16 * 512)   // 262144 halves per u matrix
#define OFF_WZ 0L
#define OFF_UZ (OFF_WZ + WFRAG)      // 65536
#define OFF_WR (OFF_UZ + UFRAG)      // 327680
#define OFF_UR (OFF_WR + WFRAG)      // 393216
#define OFF_WH (OFF_UR + UFRAG)      // 655360
#define OFF_UH (OFF_WH + WFRAG)      // 720896
#define WS_TOTAL (OFF_UH + UFRAG)    // 983040 halves = 1.92 MB

__device__ __forceinline__ unsigned short f2bf(float f) {
    union { float f; unsigned u; } c; c.f = f;
    unsigned r = c.u + 0x7FFFu + ((c.u >> 16) & 1u);   // round-to-nearest-even
    return (unsigned short)(r >> 16);
}
__device__ __forceinline__ float bf2f(unsigned short b) {
    union { unsigned u; float f; } c; c.u = ((unsigned)b) << 16;
    return c.f;
}

union FragU { uint4 q[2]; v16bf v; };

// A fragment (16xK slice, row-major bf16 in LDS, stride S halves), per ISA A layout:
// lanes 0-15: M=lane, K = kbase+{0..7, 16..23}; lanes 16-31: M=lane-16, K = kbase+{8..15, 24..31}
__device__ __forceinline__ v16bf load_afrag(const unsigned short* base, int S, int kbase, int lane) {
    int m    = lane & 15;
    int koff = (lane >> 4) << 3;           // 0 or 8
    FragU u;
    u.q[0] = *(const uint4*)(base + m * S + kbase + koff);
    u.q[1] = *(const uint4*)(base + m * S + kbase + 16 + koff);
    return u.v;
}

// B fragment: pre-swizzled in workspace, 16 contiguous halves per lane.
__device__ __forceinline__ v16bf load_bfrag(const unsigned short* ws, long fragbase, int lane) {
    const unsigned short* p = ws + fragbase + lane * 16;
    FragU u;
    u.q[0] = ((const uint4*)p)[0];
    u.q[1] = ((const uint4*)p)[1];
    return u.v;
}

__device__ __forceinline__ v8f wmma_bf16(v16bf a, v16bf b, v8f c) {
    return __builtin_amdgcn_wmma_f32_16x16x32_bf16(false, a, false, b, (short)0, c, false, false);
}

// ---------------- prep: swizzle f32 weights -> bf16 B-fragment layout in ws -----------
// B-fragment lane layout (32x16 K-major tile): lane n<16 -> N=n, K=kbase+{0..7, 16..23} (elems 0..15)
//                                              lane n>=16 -> N=n-16, K=kbase+{8..15, 24..31}
__global__ __launch_bounds__(256) void GRUDECAY_prep(
    const float* __restrict__ wz, const float* __restrict__ uz,
    const float* __restrict__ wr, const float* __restrict__ ur,
    const float* __restrict__ wh, const float* __restrict__ uh,
    unsigned short* __restrict__ ws)
{
    long idx = (long)blockIdx.x * 256 + threadIdx.x;
    if (idx >= WS_TOTAL) return;
    const float* src; int nks; long rem;
    if      (idx < OFF_UZ) { src = wz; nks = 4;  rem = idx - OFF_WZ; }
    else if (idx < OFF_WR) { src = uz; nks = 16; rem = idx - OFF_UZ; }
    else if (idx < OFF_UR) { src = wr; nks = 4;  rem = idx - OFF_WR; }
    else if (idx < OFF_WH) { src = ur; nks = 16; rem = idx - OFF_UR; }
    else if (idx < OFF_UH) { src = wh; nks = 4;  rem = idx - OFF_WH; }
    else                   { src = uh; nks = 16; rem = idx - OFF_UH; }
    int ntile = (int)(rem / (nks * 512L));
    int r2    = (int)(rem % (nks * 512L));
    int kstep = r2 / 512;
    int e     = r2 % 512;
    int lane  = e / 16;
    int j     = e % 16;
    int col   = ntile * 16 + (lane & 15);
    int k     = kstep * 32 + ((lane < 16) ? 0 : 8) + j + ((j < 8) ? 0 : 8);
    ws[idx] = f2bf(src[(long)k * Usz + col]);
}

// ---------------- main persistent GRU scan ----------------
__global__ __launch_bounds__(256) void GRUDECAY_scan(
    const float* __restrict__ inputs,   // (B, T, F+1)
    const float* __restrict__ h0,       // (B, U)
    const float* __restrict__ w_int,    // (1, U)
    const float* __restrict__ b_z, const float* __restrict__ b_r,
    const float* __restrict__ b_h, const float* __restrict__ b_int,
    const unsigned short* __restrict__ wsw,
    float* __restrict__ out)            // (B, T, U)
{
    __shared__ alignas(16) unsigned short hbuf[2][16 * Usz];  // h double-buffer (bf16)
    __shared__ alignas(16) unsigned short hrbuf[16 * Usz];    // h*r (bf16)
    __shared__ alignas(16) unsigned short xbuf[16 * Fsz];     // x (bf16)
    __shared__ float xint[16];
    __shared__ float wint_s[Usz];

    const int tid  = threadIdx.x;
    const int lane = tid & 31;
    const int wave = tid >> 5;           // 8 waves, wave owns cols [wave*64, wave*64+64)
    const int b0   = blockIdx.x * 16;    // 16 batch rows per WG

    const float bzv = b_z[0], brv = b_r[0], bhv = b_h[0], biv = b_int[0];

    for (int i = tid; i < Usz; i += 256) wint_s[i] = w_int[i];
    for (int i = tid; i < 16 * Usz; i += 256)
        hbuf[0][i] = f2bf(h0[(long)(b0 + (i >> 9)) * Usz + (i & 511)]);

    int cur = 0;
    for (int t = 0; t < Tsz; ++t) {
        // stage x_t (16 x 129 f32) -> bf16 LDS + interval column
        for (int i = tid; i < 16 * (Fsz + 1); i += 256) {
            int m = i / (Fsz + 1), f = i % (Fsz + 1);
            float v = inputs[((long)(b0 + m) * Tsz + t) * (Fsz + 1) + f];
            if (f < Fsz) xbuf[m * Fsz + f] = f2bf(v);
            else         xint[m] = v;
        }
        __syncthreads();

        const unsigned short* hcur = hbuf[cur];
        unsigned short*       hnxt = hbuf[cur ^ 1];
        v8f zreg[4], ahx[4];

        // ---- pass 1: z, r gates + x @ w_h partial; write h*r ----
        for (int it = 0; it < 4; ++it) {
            const int ntile = wave * 4 + it;
            const long wb = (long)ntile * 4;
            const long ub = (long)ntile * 16;
            v8f az = {}, ar = {}, ah = {};

            // x @ {w_z, w_r, w_h}: 4 K-steps, 3 streams, prefetch one step ahead
            {
                v16bf bz0 = load_bfrag(wsw, OFF_WZ + wb * 512, lane);
                v16bf br0 = load_bfrag(wsw, OFF_WR + wb * 512, lane);
                v16bf bh0 = load_bfrag(wsw, OFF_WH + wb * 512, lane);
                #pragma unroll
                for (int ks = 0; ks < Fsz / 32; ++ks) {
                    v16bf a = load_afrag(xbuf, Fsz, ks * 32, lane);
                    v16bf bz1 = bz0, br1 = br0, bh1 = bh0;
                    if (ks + 1 < Fsz / 32) {
                        bz1 = load_bfrag(wsw, OFF_WZ + (wb + ks + 1) * 512, lane);
                        br1 = load_bfrag(wsw, OFF_WR + (wb + ks + 1) * 512, lane);
                        bh1 = load_bfrag(wsw, OFF_WH + (wb + ks + 1) * 512, lane);
                    }
                    az = wmma_bf16(a, bz0, az);
                    ar = wmma_bf16(a, br0, ar);
                    ah = wmma_bf16(a, bh0, ah);
                    bz0 = bz1; br0 = br1; bh0 = bh1;
                }
            }
            // h @ {u_z, u_r}: 16 K-steps, 2 streams, prefetch one step ahead
            {
                v16bf bz0 = load_bfrag(wsw, OFF_UZ + ub * 512, lane);
                v16bf br0 = load_bfrag(wsw, OFF_UR + ub * 512, lane);
                #pragma unroll
                for (int ks = 0; ks < Usz / 32; ++ks) {
                    v16bf a = load_afrag(hcur, Usz, ks * 32, lane);
                    v16bf bz1 = bz0, br1 = br0;
                    if (ks + 1 < Usz / 32) {
                        bz1 = load_bfrag(wsw, OFF_UZ + (ub + ks + 1) * 512, lane);
                        br1 = load_bfrag(wsw, OFF_UR + (ub + ks + 1) * 512, lane);
                    }
                    az = wmma_bf16(a, bz0, az);
                    ar = wmma_bf16(a, br0, ar);
                    bz0 = bz1; br0 = br1;
                }
            }
            const int col   = ntile * 16 + (lane & 15);
            const int mbase = (lane < 16) ? 0 : 8;
            v8f z;
            #pragma unroll
            for (int i = 0; i < 8; ++i) {
                z[i]     = 1.f / (1.f + __expf(-(az[i] + bzv)));
                float rv = 1.f / (1.f + __expf(-(ar[i] + brv)));
                int m    = mbase + i;
                float ho = bf2f(hcur[m * Usz + col]);
                hrbuf[m * Usz + col] = f2bf(ho * rv);
            }
            zreg[it] = z;
            ahx[it]  = ah;
        }
        __syncthreads();   // h*r complete across all waves

        // ---- pass 2: (h*r) @ u_h, tanh, decay, emit ----
        for (int it = 0; it < 4; ++it) {
            const int ntile = wave * 4 + it;
            const long ub = (long)ntile * 16;
            v8f ah = ahx[it];
            {
                v16bf bh0 = load_bfrag(wsw, OFF_UH + ub * 512, lane);
                #pragma unroll
                for (int ks = 0; ks < Usz / 32; ++ks) {
                    v16bf a = load_afrag(hrbuf, Usz, ks * 32, lane);
                    v16bf bh1 = bh0;
                    if (ks + 1 < Usz / 32)
                        bh1 = load_bfrag(wsw, OFF_UH + (ub + ks + 1) * 512, lane);
                    ah = wmma_bf16(a, bh0, ah);
                    bh0 = bh1;
                }
            }
            const int col   = ntile * 16 + (lane & 15);
            const int mbase = (lane < 16) ? 0 : 8;
            v8f z = zreg[it];
            #pragma unroll
            for (int i = 0; i < 8; ++i) {
                int   m  = mbase + i;
                float hh = tanhf(ah[i] + bhv);
                float ho = bf2f(hcur[m * Usz + col]);
                float d  = __expf(-fmaxf(0.f, xint[m] * wint_s[col] + biv));
                float hn = d * (z[i] * ho + (1.f - z[i]) * hh);
                out[((long)(b0 + m) * Tsz + t) * Usz + col] = hn;
                hnxt[m * Usz + col] = f2bf(hn);
            }
        }
        __syncthreads();   // h_{t+1} ready, xbuf free for reuse
        cur ^= 1;
    }
}

extern "C" void kernel_launch(void* const* d_in, const int* in_sizes, int n_in,
                              void* d_out, int out_size, void* d_ws, size_t ws_size,
                              hipStream_t stream) {
    const float* inputs = (const float*)d_in[0];
    const float* h0     = (const float*)d_in[1];
    const float* w_z    = (const float*)d_in[2];
    const float* u_z    = (const float*)d_in[3];
    const float* b_z    = (const float*)d_in[4];
    const float* w_r    = (const float*)d_in[5];
    const float* u_r    = (const float*)d_in[6];
    const float* b_r    = (const float*)d_in[7];
    const float* w_h    = (const float*)d_in[8];
    const float* u_h    = (const float*)d_in[9];
    const float* b_h    = (const float*)d_in[10];
    const float* w_int  = (const float*)d_in[11];
    const float* b_int  = (const float*)d_in[12];
    unsigned short* wsb = (unsigned short*)d_ws;
    float* out          = (float*)d_out;

    GRUDECAY_prep<<<(int)((WS_TOTAL + 255) / 256), 256, 0, stream>>>(
        w_z, u_z, w_r, u_r, w_h, u_h, wsb);
    GRUDECAY_scan<<<Bsz / 16, 256, 0, stream>>>(
        inputs, h0, w_int, b_z, b_r, b_h, b_int, wsb, out);
}
